// Res16UNet34C_21492016349520
// MI455X (gfx1250) — matrix operand order repaired and checked by reading, not compiled, not run on hardware
//
#include <hip/hip_runtime.h>
#include <stdint.h>

// ---------------------------------------------------------------------------
// Res16UNet sparse-conv forward for gfx1250 (MI455X).
// bf16 activations + transposed bf16 weights, WMMA f32_16x16x32_bf16,
// fp32 accumulation + fp32 BN stats, fp32 atomic scatter.
// A-fragment: two contiguous b128 loads/lane; B-fragment: two contiguous
// b128 loads/lane from (K,Cout,Cin)-transposed weights; zero-row clamping
// keeps control flow uniform (EXEC all-ones for WMMA).
// ---------------------------------------------------------------------------

typedef __attribute__((ext_vector_type(16))) __bf16         bf16x16;
typedef __attribute__((ext_vector_type(8)))  unsigned short u16x8;
typedef __attribute__((ext_vector_type(8)))  float          v8f;

struct U16pair { u16x8 lo, hi; };

__device__ __forceinline__ unsigned short f2bfu(float f) {
  unsigned u = __builtin_bit_cast(unsigned, f);
  return (unsigned short)((u + 0x7FFFu + ((u >> 16) & 1u)) >> 16);
}
__device__ __forceinline__ float bf2f(unsigned short b) {
  return __builtin_bit_cast(float, (unsigned)b << 16);
}
__device__ __forceinline__ bf16x16 mk16(u16x8 lo, u16x8 hi) {
  U16pair p{lo, hi};
  return __builtin_bit_cast(bf16x16, p);
}

// --------------------------- elementwise kernels ---------------------------

__global__ void k_zero_h(unsigned short* __restrict__ p, long long n) {
  for (long long i = blockIdx.x * (long long)blockDim.x + threadIdx.x; i < n;
       i += (long long)gridDim.x * blockDim.x)
    p[i] = 0;
}

// feats (n0,3) f32 -> (n0,32) bf16 zero-padded
__global__ void k_pad_feats(const float* __restrict__ src,
                            unsigned short* __restrict__ dst, long long n0) {
  long long total = n0 * 32;
  for (long long i = blockIdx.x * (long long)blockDim.x + threadIdx.x; i < total;
       i += (long long)gridDim.x * blockDim.x) {
    int c = (int)(i & 31);
    long long r = i >> 5;
    dst[i] = (c < 3) ? f2bfu(src[r * 3 + c]) : (unsigned short)0;
  }
}

// W (K,Cin,Cout) f32 -> WT (K,Cout,CinP) bf16, Cin zero-padded to CinP
__global__ void k_cvt_wT(const float* __restrict__ src,
                         unsigned short* __restrict__ dst,
                         int K, int Cin, int Cout, int CinP) {
  long long total = (long long)K * Cout * CinP;
  for (long long i = blockIdx.x * (long long)blockDim.x + threadIdx.x; i < total;
       i += (long long)gridDim.x * blockDim.x) {
    int ci = (int)(i % CinP);
    long long t = i / CinP;
    int co = (int)(t % Cout);
    int k  = (int)(t / Cout);
    dst[i] = (ci < Cin) ? f2bfu(src[((long long)k * Cin + ci) * Cout + co])
                        : (unsigned short)0;
  }
}

__global__ void k_zero(float* __restrict__ p, const int* __restrict__ np, int C) {
  long long total = (long long)(*np) * C;
  for (long long i = blockIdx.x * (long long)blockDim.x + threadIdx.x; i < total;
       i += (long long)gridDim.x * blockDim.x)
    p[i] = 0.f;
}

__global__ void k_bnapply(const float* __restrict__ x, const float* __restrict__ mean,
                          const float* __restrict__ rstd,
                          const unsigned short* __restrict__ res,
                          unsigned short* __restrict__ out,
                          const int* __restrict__ np, int C, int relu) {
  long long total = (long long)(*np) * C;
  for (long long i = blockIdx.x * (long long)blockDim.x + threadIdx.x; i < total;
       i += (long long)gridDim.x * blockDim.x) {
    int c = (int)(i % C);
    float v = (x[i] - mean[c]) * rstd[c];
    if (res) v += bf2f(res[i]);
    if (relu) v = fmaxf(v, 0.f);
    out[i] = f2bfu(v);
  }
}

__global__ void k_bnstats(const float* __restrict__ x, const int* __restrict__ np,
                          int C, float* __restrict__ mean, float* __restrict__ rstd) {
  const int c = blockIdx.x;
  const int n = *np;
  float s = 0.f, s2 = 0.f;
  for (int i = threadIdx.x; i < n; i += blockDim.x) {
    float v = x[(long long)i * C + c];
    s += v; s2 += v * v;
  }
  __shared__ float sh[256], sh2[256];
  sh[threadIdx.x] = s; sh2[threadIdx.x] = s2;
  __syncthreads();
  for (int o = 128; o > 0; o >>= 1) {
    if ((int)threadIdx.x < o) { sh[threadIdx.x] += sh[threadIdx.x + o];
                                sh2[threadIdx.x] += sh2[threadIdx.x + o]; }
    __syncthreads();
  }
  if (threadIdx.x == 0) {
    float m = sh[0] / (float)n;
    float v = sh2[0] / (float)n - m * m;
    mean[c] = m;
    rstd[c] = rsqrtf(fmaxf(v, 0.f) + 1e-5f);
  }
}

__global__ void k_concat(const unsigned short* __restrict__ a, int Ca,
                         const unsigned short* __restrict__ b, int Cb,
                         unsigned short* __restrict__ out, const int* __restrict__ np) {
  const int C = Ca + Cb;
  long long total = (long long)(*np) * C;
  for (long long i = blockIdx.x * (long long)blockDim.x + threadIdx.x; i < total;
       i += (long long)gridDim.x * blockDim.x) {
    long long r = i / C; int c = (int)(i % C);
    out[i] = (c < Ca) ? a[r * Ca + c] : b[r * Cb + (c - Ca)];
  }
}

__global__ void k_emit(float* __restrict__ dst,
                       const unsigned short* __restrict__ x4,
                       const unsigned short* __restrict__ d5,
                       const unsigned short* __restrict__ d6,
                       const unsigned short* __restrict__ d7,
                       const unsigned short* __restrict__ d8,
                       const int* n0p, const int* n1p, const int* n2p,
                       const int* n3p, const int* n4p) {
  long long n0 = *n0p, n1 = *n1p, n2 = *n2p, n3 = *n3p, n4 = *n4p;
  long long o1 = n4 * 256;
  long long o2 = o1 + n3 * 256;
  long long o3 = o2 + n2 * 128;
  long long o4 = o3 + n1 * 96;
  long long total = o4 + n0 * 96;
  for (long long i = blockIdx.x * (long long)blockDim.x + threadIdx.x; i < total;
       i += (long long)gridDim.x * blockDim.x) {
    float v;
    if      (i < o1) v = bf2f(x4[i]);
    else if (i < o2) v = bf2f(d5[i - o1]);
    else if (i < o3) v = bf2f(d6[i - o2]);
    else if (i < o4) v = bf2f(d7[i - o3]);
    else             v = bf2f(d8[i - o4]);
    dst[i] = v;
  }
}

// ------------------------ WMMA gather-GEMM-scatter -------------------------
// One wave32 per (16 pairs) x (16*NT out channels) tile of tap k.
// Cin must be a multiple of 32. A loads: 2x b128/lane; B loads: 2x b128/lane
// per tile from transposed weights. Dummy rows read from zrow (uniform flow).
template <int NT>
__global__ __launch_bounds__(32) void k_spconv(
    const unsigned short* __restrict__ feat,   // (n_in, Cin) bf16
    const unsigned short* __restrict__ WT,     // (K, Cout, Cin) bf16
    const unsigned short* __restrict__ zrow,   // >= Cin zeros
    const int* __restrict__ gidx,              // (K, P)
    const int* __restrict__ sidx,              // (K, P)
    float* __restrict__ out,                   // (n_out, Cout) f32 accum
    const int* __restrict__ nin_p, const int* __restrict__ nout_p,
    int P, int Cin, int Cout) {
  const int k  = blockIdx.z;
  const int p0 = blockIdx.x << 4;
  const int c0 = blockIdx.y * (16 * NT);
  const int lane = threadIdx.x;
  const int hf = lane >> 4, m = lane & 15;
  const int nin  = *nin_p;
  const int nout = *nout_p;
  const int* gk = gidx + (long long)k * P;
  const int* sk = sidx + (long long)k * P;
  const unsigned short* arow = zrow;
  if (p0 + m < P) {
    int idx = gk[p0 + m];
    if (idx < nin) arow = feat + (long long)idx * Cin;
  }
  const unsigned short* a0 = arow + (hf << 4);
  const unsigned short* b0 =
      WT + ((long long)k * Cout + c0 + m) * Cin + (hf << 4);
  v8f acc[NT] = {};
  for (int kc = 0; kc < Cin; kc += 32) {
    bf16x16 A = mk16(*(const u16x8*)(a0 + kc), *(const u16x8*)(a0 + kc + 8));
#pragma unroll
    for (int t = 0; t < NT; ++t) {
      const unsigned short* bt = b0 + (long long)t * 16 * Cin + kc;
      bf16x16 B = mk16(*(const u16x8*)(bt), *(const u16x8*)(bt + 8));
      acc[t] = __builtin_amdgcn_wmma_f32_16x16x32_bf16(
          false, A, false, B, (short)0, acc[t], false, false);
    }
  }
#pragma unroll
  for (int r = 0; r < 8; ++r) {
    int pm = p0 + (hf << 3) + r;   // D fragment: M = 8*half + vgpr
    if (pm < P) {
      int orow = sk[pm];
      if (orow < nout) {
        float* ob = out + (long long)orow * Cout + c0 + m;
#pragma unroll
        for (int t = 0; t < NT; ++t) atomicAdd(ob + t * 16, acc[t][r]);
      }
    }
  }
}

// Dense GEMM out(n,Cout) = X(n,Cin) @ W; WT is (Cout,Cin) bf16; NT=2 tiles.
__global__ __launch_bounds__(32) void k_gemm(
    const unsigned short* __restrict__ X, const unsigned short* __restrict__ WT,
    const unsigned short* __restrict__ zrow,
    float* __restrict__ out, const int* __restrict__ n_p, int Cin, int Cout) {
  const int n = *n_p;
  const int lane = threadIdx.x, hf = lane >> 4, m = lane & 15;
  const int nct = Cout >> 5;
  const int ntiles = ((n + 15) >> 4) * nct;
  for (int t = blockIdx.x; t < ntiles; t += gridDim.x) {
    const int rt = t / nct, ct = t % nct;
    const int row = (rt << 4) + m;
    const unsigned short* a0 =
        ((row < n) ? X + (long long)row * Cin : zrow) + (hf << 4);
    const unsigned short* b0 =
        WT + (long long)((ct << 5) + m) * Cin + (hf << 4);
    v8f acc0 = {}, acc1 = {};
    for (int kc = 0; kc < Cin; kc += 32) {
      bf16x16 A = mk16(*(const u16x8*)(a0 + kc), *(const u16x8*)(a0 + kc + 8));
      bf16x16 B0 = mk16(*(const u16x8*)(b0 + kc), *(const u16x8*)(b0 + kc + 8));
      const unsigned short* b1 = b0 + (long long)16 * Cin;
      bf16x16 B1 = mk16(*(const u16x8*)(b1 + kc), *(const u16x8*)(b1 + kc + 8));
      acc0 = __builtin_amdgcn_wmma_f32_16x16x32_bf16(false, A, false, B0,
                                                     (short)0, acc0, false, false);
      acc1 = __builtin_amdgcn_wmma_f32_16x16x32_bf16(false, A, false, B1,
                                                     (short)0, acc1, false, false);
    }
#pragma unroll
    for (int r = 0; r < 8; ++r) {
      int rr = (rt << 4) + (hf << 3) + r;
      if (rr < n) {
        float* ob = out + (long long)rr * Cout + (ct << 5) + m;
        ob[0]  = acc0[r];
        ob[16] = acc1[r];
      }
    }
  }
}

// ------------------------------- host side ---------------------------------

static inline int idivup(int a, int b) { return (a + b - 1) / b; }

extern "C" void kernel_launch(void* const* d_in, const int* in_sizes, int n_in,
                              void* d_out, int out_size, void* d_ws, size_t ws_size,
                              hipStream_t stream) {
  (void)n_in; (void)out_size; (void)ws_size;

  // ---- input map (setup_inputs() insertion-order flatten) ----
  const int* s0i = (const int*)d_in[63]; const int* s0o = (const int*)d_in[64];
  const int* s1i = (const int*)d_in[65]; const int* s1o = (const int*)d_in[66];
  const int* s2i = (const int*)d_in[67]; const int* s2o = (const int*)d_in[68];
  const int* s3i = (const int*)d_in[69]; const int* s3o = (const int*)d_in[70];
  const int* s4i = (const int*)d_in[71]; const int* s4o = (const int*)d_in[72];
  const int* d1i = (const int*)d_in[73]; const int* d1o = (const int*)d_in[74];
  const int* d2i = (const int*)d_in[75]; const int* d2o = (const int*)d_in[76];
  const int* d3i = (const int*)d_in[77]; const int* d3o = (const int*)d_in[78];
  const int* d4i = (const int*)d_in[79]; const int* d4o = (const int*)d_in[80];
  const int* n0p = (const int*)d_in[81]; const int* n1p = (const int*)d_in[82];
  const int* n2p = (const int*)d_in[83]; const int* n3p = (const int*)d_in[84];
  const int* n4p = (const int*)d_in[85];

  const int P0 = in_sizes[63] / 27, P1 = in_sizes[65] / 27, P2 = in_sizes[67] / 27;
  const int P3 = in_sizes[69] / 27, P4 = in_sizes[71] / 27;
  const int Q1 = in_sizes[73] / 8,  Q2 = in_sizes[75] / 8;
  const int Q3 = in_sizes[77] / 8,  Q4 = in_sizes[79] / 8;

  const size_t NB = (size_t)(in_sizes[0] / 3);   // n0 (upper bound for all levels)

  // ---- workspace layout (d_ws is 256B-aligned; keep 16B alignment) ----
  char* wsp = (char*)d_ws;
  auto alloc_f = [&](size_t n) -> float* {
    float* r = (float*)wsp; wsp += ((n + 3) & ~(size_t)3) * sizeof(float); return r; };
  auto alloc_h = [&](size_t n) -> unsigned short* {
    unsigned short* r = (unsigned short*)wsp;
    wsp += ((n + 7) & ~(size_t)7) * sizeof(unsigned short); return r; };

  float* tmp1  = alloc_f(NB * 256);   // fp32 conv accumulator
  float* resb  = alloc_f(NB * 256);   // fp32 wd-gemm output
  float* mean  = alloc_f(512); float* rstd  = alloc_f(512);
  float* mean2 = alloc_f(512); float* rstd2 = alloc_f(512);

  unsigned short* x0   = alloc_h(NB * 32);
  unsigned short* x1   = alloc_h(NB * 32);
  unsigned short* x2   = alloc_h(NB * 64);
  unsigned short* x3   = alloc_h(NB * 128);
  unsigned short* x4   = alloc_h(NB * 256);
  unsigned short* d5   = alloc_h(NB * 256);
  unsigned short* d6   = alloc_h(NB * 128);
  unsigned short* d7   = alloc_h(NB * 96);
  unsigned short* d8   = alloc_h(NB * 96);
  unsigned short* stg  = alloc_h(NB * 128);
  unsigned short* hbuf = alloc_h(NB * 256);
  unsigned short* resh = alloc_h(NB * 256);
  unsigned short* bufA = alloc_h(NB * 384);
  unsigned short* upb  = alloc_h(NB * 256);
  unsigned short* fbf  = alloc_h(NB * 32);               // padded bf16 feats
  unsigned short* zrow = alloc_h(512);                   // shared zero row
  unsigned short* wtb  = alloc_h((size_t)27 * 384 * 256); // transposed-weight scratch

  // ---- init: zero row + padded feats ----
  k_zero_h<<<2, 256, 0, stream>>>(zrow, 512);
  {
    long long n = (long long)NB * 32;
    int g = (int)((n + 255) / 256); if (g > 4096) g = 4096;
    k_pad_feats<<<g, 256, 0, stream>>>((const float*)d_in[0], fbf, (long long)NB);
  }

  // ---- helpers ----
  auto conv = [&](const unsigned short* f, const float* W, int K, int Cin, int Cout,
                  const int* gi, const int* si, float* acc,
                  const int* ninp, const int* noutp, int P) {
    int CinP = (Cin % 32) ? ((Cin + 31) / 32) * 32 : Cin;
    long long wn = (long long)K * Cout * CinP;
    int gw = (int)((wn + 255) / 256); if (gw > 4096) gw = 4096;
    k_cvt_wT<<<gw, 256, 0, stream>>>(W, wtb, K, Cin, Cout, CinP);
    k_zero<<<2048, 256, 0, stream>>>(acc, noutp, Cout);
    int NT = (Cout % 64 == 0) ? 4 : 2;     // all Couts are multiples of 32
    dim3 g(idivup(P, 16), Cout / (16 * NT), K);
    if (NT == 4)
      k_spconv<4><<<g, 32, 0, stream>>>(f, wtb, zrow, gi, si, acc, ninp, noutp,
                                        P, CinP, Cout);
    else
      k_spconv<2><<<g, 32, 0, stream>>>(f, wtb, zrow, gi, si, acc, ninp, noutp,
                                        P, CinP, Cout);
  };
  auto bnact = [&](float* x, const int* np, int C, const unsigned short* res,
                   unsigned short* out, int relu, float* mb, float* rb) {
    k_bnstats<<<C, 256, 0, stream>>>(x, np, C, mb, rb);
    k_bnapply<<<2048, 256, 0, stream>>>(x, mb, rb, res, out, np, C, relu);
  };
  auto block = [&](const unsigned short* fin, int Ci, unsigned short* fout, int Co,
                   const float* w1, const float* w2, const float* wd,
                   const int* ri, const int* ro, int P, const int* np) {
    conv(fin, w1, 27, Ci, Co, ri, ro, tmp1, np, np, P);
    bnact(tmp1, np, Co, nullptr, hbuf, 1, mean, rstd);           // h = relu(bn(conv1))
    conv(hbuf, w2, 27, Co, Co, ri, ro, tmp1, np, np, P);
    k_bnstats<<<Co, 256, 0, stream>>>(tmp1, np, Co, mean, rstd); // bn2 stats
    const unsigned short* res = fin;
    if (wd) {
      long long wn = (long long)Ci * Co;
      int gw = (int)((wn + 255) / 256); if (gw > 4096) gw = 4096;
      k_cvt_wT<<<gw, 256, 0, stream>>>(wd, wtb, 1, Ci, Co, Ci);
      k_gemm<<<8192, 32, 0, stream>>>(fin, wtb, zrow, resb, np, Ci, Co);
      bnact(resb, np, Co, nullptr, resh, 0, mean2, rstd2);       // res = bn(f@wd)
      res = resh;
    }
    k_bnapply<<<2048, 256, 0, stream>>>(tmp1, mean, rstd, res, fout, np, Co, 1);
  };
  auto F = [&](int i) { return (const float*)d_in[i]; };

  // ---- encoder ----
  conv(fbf, F(1), 27, 3, 32, s0i, s0o, tmp1, n0p, n0p, P0);
  bnact(tmp1, n0p, 32, nullptr, x0, 1, mean, rstd);

  conv(x0, F(2), 8, 32, 32, d1i, d1o, tmp1, n0p, n1p, Q1);
  bnact(tmp1, n1p, 32, nullptr, x1, 1, mean, rstd);
  block(x1, 32, x1, 32, F(3), F(4), nullptr, s1i, s1o, P1, n1p);
  block(x1, 32, x1, 32, F(5), F(6), nullptr, s1i, s1o, P1, n1p);

  conv(x1, F(7), 8, 32, 32, d2i, d2o, tmp1, n1p, n2p, Q2);
  bnact(tmp1, n2p, 32, nullptr, stg, 1, mean, rstd);
  block(stg, 32, x2, 64, F(8),  F(9),  F(10),  s2i, s2o, P2, n2p);
  block(x2,  64, x2, 64, F(11), F(12), nullptr, s2i, s2o, P2, n2p);
  block(x2,  64, x2, 64, F(13), F(14), nullptr, s2i, s2o, P2, n2p);

  conv(x2, F(15), 8, 64, 64, d3i, d3o, tmp1, n2p, n3p, Q3);
  bnact(tmp1, n3p, 64, nullptr, stg, 1, mean, rstd);
  block(stg, 64,  x3, 128, F(16), F(17), F(18),  s3i, s3o, P3, n3p);
  block(x3,  128, x3, 128, F(19), F(20), nullptr, s3i, s3o, P3, n3p);
  block(x3,  128, x3, 128, F(21), F(22), nullptr, s3i, s3o, P3, n3p);
  block(x3,  128, x3, 128, F(23), F(24), nullptr, s3i, s3o, P3, n3p);

  conv(x3, F(25), 8, 128, 128, d4i, d4o, tmp1, n3p, n4p, Q4);
  bnact(tmp1, n4p, 128, nullptr, stg, 1, mean, rstd);
  block(stg, 128, x4, 256, F(26), F(27), F(28),  s4i, s4o, P4, n4p);
  block(x4,  256, x4, 256, F(29), F(30), nullptr, s4i, s4o, P4, n4p);
  block(x4,  256, x4, 256, F(31), F(32), nullptr, s4i, s4o, P4, n4p);
  block(x4,  256, x4, 256, F(33), F(34), nullptr, s4i, s4o, P4, n4p);
  block(x4,  256, x4, 256, F(35), F(36), nullptr, s4i, s4o, P4, n4p);
  block(x4,  256, x4, 256, F(37), F(38), nullptr, s4i, s4o, P4, n4p);

  // ---- decoder ----
  auto uplev = [&](const unsigned short* src, int Cc, const int* cnp,
                   const float* upW, int Cu,
                   const int* di, const int* dO, int Q, const int* fnp,
                   const unsigned short* skip, int Cs,
                   const int* ri, const int* ro, int P,
                   const float* w01, const float* w02, const float* w0d,
                   const float* w11, const float* w12,
                   unsigned short* outb, int Co) {
    // inverse conv: gather coarse (dO, dummy==n_coarse), scatter fine (di)
    conv(src, upW, 8, Cc, Cu, dO, di, tmp1, cnp, fnp, Q);
    bnact(tmp1, fnp, Cu, nullptr, upb, 1, mean, rstd);
    k_concat<<<2048, 256, 0, stream>>>(upb, Cu, skip, Cs, bufA, fnp);
    block(bufA, Cu + Cs, outb, Co, w01, w02, w0d, ri, ro, P, fnp);
    block(outb, Co,      outb, Co, w11, w12, nullptr, ri, ro, P, fnp);
  };

  uplev(x4, 256, n4p, F(39), 256, d4i, d4o, Q4, n3p, x3, 128, s3i, s3o, P3,
        F(40), F(41), F(42), F(43), F(44), d5, 256);
  uplev(d5, 256, n3p, F(45), 128, d3i, d3o, Q3, n2p, x2, 64,  s2i, s2o, P2,
        F(46), F(47), F(48), F(49), F(50), d6, 128);
  uplev(d6, 128, n2p, F(51), 96,  d2i, d2o, Q2, n1p, x1, 32,  s1i, s1o, P1,
        F(52), F(53), F(54), F(55), F(56), d7, 96);
  uplev(d7, 96,  n1p, F(57), 96,  d1i, d1o, Q1, n0p, x0, 32,  s0i, s0o, P0,
        F(58), F(59), F(60), F(61), F(62), d8, 96);

  // ---- emit tuple (x4, d5, d6, d7, d8) as f32 at device-computed offsets ----
  k_emit<<<4096, 256, 0, stream>>>((float*)d_out, x4, d5, d6, d7, d8,
                                   n0p, n1p, n2p, n3p, n4p);
}
